// M1OptimizedMultiHeadAttention_37203006718304
// MI455X (gfx1250) — matrix operand order repaired and checked by reading, not compiled, not run on hardware
//
#include <hip/hip_runtime.h>

typedef unsigned short u16;
typedef __attribute__((ext_vector_type(8)))  u16    u16x8;
typedef __attribute__((ext_vector_type(16))) u16    u16x16;
typedef __attribute__((ext_vector_type(16))) __bf16 v16bf;
typedef __attribute__((ext_vector_type(8)))  float  v8f;

#define BATCH   2
#define S_LEN   2048
#define DMODEL  1024
#define NHEADS  16
#define DK      64

union BfPack { u16x16 u; v16bf b; };

__device__ __forceinline__ u16 f2bf(float f) {
  unsigned int u = __float_as_uint(f);
  u += 0x7FFFu + ((u >> 16) & 1u);          // round to nearest even
  return (u16)(u >> 16);
}

__device__ __forceinline__ v8f wmma_bf16(v16bf a, v16bf b, v8f c) {
  // D = A(16x32 bf16) * B(32x16 bf16) + C(16x16 f32)
  return __builtin_amdgcn_wmma_f32_16x16x32_bf16(false, a, false, b, (short)0, c,
                                                 false, false);
}

// Async DMA: 16 bytes per lane, global -> LDS, tracked by ASYNCcnt.
__device__ __forceinline__ void async_copy_b128(u16* lds_dst, const u16* gsrc) {
  unsigned loff = (unsigned)(uintptr_t)lds_dst;   // low 32 bits = LDS offset
  asm volatile("global_load_async_to_lds_b128 %0, %1, off"
               :: "v"(loff), "v"(gsrc) : "memory");
}
__device__ __forceinline__ void wait_async0() {
  asm volatile("s_wait_asynccnt 0x0" ::: "memory");
}
__device__ __forceinline__ void wait_ds0() {
  asm volatile("s_wait_dscnt 0x0" ::: "memory");
}

// A tile 16x32 (MxK): element (m,k) read from row-major p[(m0+m)*ld + k0 + k].
// Layout: lane l holds m=l&15; half h holds k=(h&7)+((h>>3)<<4)+((l>>4)<<3).
__device__ __forceinline__ v16bf load_a16x32(const u16* p, int m0, int k0,
                                             int ld, int lane) {
  int m = lane & 15, g = lane >> 4;
  const u16* base = p + (size_t)(m0 + m) * ld + k0 + g * 8;
  u16x8 lo = *(const u16x8*)base;
  u16x8 hi = *(const u16x8*)(base + 16);
  BfPack r;
#pragma unroll
  for (int i = 0; i < 8; ++i) { r.u[i] = lo[i]; r.u[8 + i] = hi[i]; }
  return r.b;
}

// B tile 32x16 (KxN): element (k,n) read from p[(n0+n)*ld + k0 + k]
// (B^T row-major).  Lane l holds n=l&15; half h holds k=(h&7)+((h>>3)<<4)+((l>>4)<<3).
// Works for global or LDS pointers (compiler picks b128 vmem/ds loads).
__device__ __forceinline__ v16bf load_b32x16(const u16* p, int n0, int k0,
                                             int ld, int lane) {
  int n = lane & 15, g = lane >> 4;
  const u16* base = p + (size_t)(n0 + n) * ld + k0 + g * 8;
  u16x8 lo = *(const u16x8*)base;
  u16x8 hi = *(const u16x8*)(base + 16);
  BfPack r;
#pragma unroll
  for (int i = 0; i < 8; ++i) { r.u[i] = lo[i]; r.u[8 + i] = hi[i]; }
  return r.b;
}

// 16(M) x 128(N) tile accumulation over K: 8 WMMA per 32-wide K step.
__device__ __forceinline__ void gemm_tile_16x128(const u16* __restrict__ A,
                                                 const u16* __restrict__ Bt,
                                                 int K, int m0, int n0, int lane,
                                                 v8f acc[8]) {
  for (int kk = 0; kk < K; kk += 32) {
    v16bf a = load_a16x32(A, m0, kk, K, lane);
#pragma unroll
    for (int c = 0; c < 8; ++c) {
      v16bf b = load_b32x16(Bt, n0 + c * 16, kk, K, lane);
      acc[c] = wmma_bf16(a, b, acc[c]);
    }
  }
}

// ---------------- prep kernels ----------------

__global__ void cast_bf16_kernel(const float* __restrict__ in,
                                 u16* __restrict__ out, int n) {
  int i = blockIdx.x * blockDim.x + threadIdx.x;
  if (i < n) out[i] = f2bf(in[i]);
}

// in: row-major [rows x cols] f32 -> out: row-major [cols x rows] bf16
__global__ void transpose_cast_kernel(const float* __restrict__ in,
                                      u16* __restrict__ out, int rows, int cols) {
  int i = blockIdx.x * blockDim.x + threadIdx.x;
  if (i < rows * cols) {
    int r = i / cols, c = i - r * cols;
    out[(size_t)c * rows + r] = f2bf(in[i]);
  }
}

// ---------------- QKV projection ----------------
// xb:[4096 x 1024] bf16, wqT:[3072 x 1024] bf16 (W_qkv^T)
// writes q,k: [B*H][S][DK] bf16 ; vt: [B*H][DK][S] bf16
__global__ __launch_bounds__(32) void qkv_gemm_kernel(
    const u16* __restrict__ xb, const u16* __restrict__ wqT,
    const float* __restrict__ bqkv, u16* __restrict__ q, u16* __restrict__ k,
    u16* __restrict__ vt) {
  int lane = threadIdx.x;
  int ntile = blockIdx.x % (3 * DMODEL / 128);   // 24
  int mtile = blockIdx.x / (3 * DMODEL / 128);
  int m0 = mtile * 16, n0 = ntile * 128;
  v8f acc[8];
#pragma unroll
  for (int c = 0; c < 8; ++c) acc[c] = 0.0f;
  gemm_tile_16x128(xb, wqT, DMODEL, m0, n0, lane, acc);

  int m = lane & 15, g = lane >> 4;
#pragma unroll
  for (int c = 0; c < 8; ++c) {
    int ncol = n0 + c * 16 + m;
    float bias = bqkv[ncol];
    int sel = ncol >> 10;          // 0:q 1:k 2:v
    int col = ncol & (DMODEL - 1);
    int h = col >> 6, d = col & (DK - 1);
#pragma unroll
    for (int r = 0; r < 8; ++r) {
      int row = m0 + r + 8 * g;            // 0..4095
      int b = row >> 11, s = row & (S_LEN - 1);
      u16 bv = f2bf(acc[c][r] + bias);
      size_t bh = (size_t)(b * NHEADS + h);
      if (sel == 0)      q[(bh * S_LEN + s) * DK + d] = bv;
      else if (sel == 1) k[(bh * S_LEN + s) * DK + d] = bv;
      else               vt[(bh * DK + d) * S_LEN + s] = bv;
    }
  }
}

// ---------------- flash attention ----------------
// 4 waves / block cover 64 query rows of one (b,h).  K/V tiles for each
// 32-key step are staged into LDS by async-to-LDS DMA with DOUBLE BUFFERING:
// while tile i is consumed by WMMA/softmax, tile i+1 is in flight.
__device__ __forceinline__ void stage_tiles(const u16* kh, const u16* vth,
                                            int k0, u16* kbuf, u16* vbuf,
                                            int tid) {
  // K tile: rows k0..k0+31 are one contiguous 4KB block
  async_copy_b128(&kbuf[tid * 8], kh + (size_t)k0 * DK + tid * 8);
  async_copy_b128(&kbuf[(tid + 128) * 8],
                  kh + (size_t)k0 * DK + (tid + 128) * 8);
  // Vt tile: 64 rows x 64B, strided by S_LEN halves
  int i0 = tid, d0 = i0 >> 2, p0 = i0 & 3;
  async_copy_b128(&vbuf[d0 * 32 + p0 * 8],
                  vth + (size_t)d0 * S_LEN + k0 + p0 * 8);
  int i1 = tid + 128, d1 = i1 >> 2, p1 = i1 & 3;
  async_copy_b128(&vbuf[d1 * 32 + p1 * 8],
                  vth + (size_t)d1 * S_LEN + k0 + p1 * 8);
}

__global__ __launch_bounds__(128) void attn_kernel(
    const u16* __restrict__ q, const u16* __restrict__ k,
    const u16* __restrict__ vt, const int* __restrict__ mask,
    u16* __restrict__ attn_out) {
  __shared__ __align__(16) u16 lds_k[2][32 * DK];     // [buf][key][d]
  __shared__ __align__(16) u16 lds_v[2][DK * 32];     // [buf][d][key]
  __shared__ __align__(16) u16 lds_p[4][16 * 32];     // per-wave P staging

  int tid  = threadIdx.x;
  int lane = tid & 31;
  int wave = tid >> 5;
  int qblk = blockIdx.x & (S_LEN / 64 - 1);           // 32 blocks per (b,h)
  int bh   = blockIdx.x >> 5;                         // 0..31
  int b    = bh >> 4;
  int h    = bh & (NHEADS - 1);
  const u16* qh  = q  + (size_t)bh * S_LEN * DK;
  const u16* kh  = k  + (size_t)bh * S_LEN * DK;
  const u16* vth = vt + (size_t)bh * DK * S_LEN;
  const int* mb  = mask + b * S_LEN;
  int m = lane & 15, g = lane >> 4;
  int q0 = qblk * 64 + wave * 16;

  // Q A-tiles (d = 0..31 and 32..63), reused across all key steps
  v16bf qa0 = load_a16x32(qh, q0, 0, DK, lane);
  v16bf qa1 = load_a16x32(qh, q0, 32, DK, lane);

  float mrow[8], lrow[8];
  v8f acc[4];
#pragma unroll
  for (int r = 0; r < 8; ++r) { mrow[r] = -1e30f; lrow[r] = 0.0f; }
#pragma unroll
  for (int c = 0; c < 4; ++c) acc[c] = 0.0f;

  // prologue: launch DMA for the first tile
  stage_tiles(kh, vth, 0, lds_k[0], lds_v[0], tid);

  for (int k0 = 0; k0 < S_LEN; k0 += 32) {
    int cur = (k0 >> 5) & 1;
    const u16* kt = lds_k[cur];
    const u16* vtile = lds_v[cur];

    wait_async0();                 // this wave's share of tile(k0) landed
    __syncthreads();               // all waves' shares landed; prior readers
                                   // of the other buffer are finished
    if (k0 + 32 < S_LEN)           // kick off tile(k0+32) into the other buf
      stage_tiles(kh, vth, k0 + 32, lds_k[cur ^ 1], lds_v[cur ^ 1], tid);

    // ---- scores S = Q K^T / sqrt(DK): two 16x16 n-tiles, K-dim 64 = 2 steps
    v8f s[2];
#pragma unroll
    for (int t = 0; t < 2; ++t) {
      v8f c = 0.0f;
      c = wmma_bf16(qa0, load_b32x16(kt, t * 16, 0, DK, lane), c);
      c = wmma_bf16(qa1, load_b32x16(kt, t * 16, 32, DK, lane), c);
      s[t] = c;
    }
    // scale + mask (mask indexed by key = n lane position)
#pragma unroll
    for (int t = 0; t < 2; ++t) {
      float madd = (mb[k0 + t * 16 + m] == 0) ? -1.0e9f : 0.0f;
#pragma unroll
      for (int r = 0; r < 8; ++r) s[t][r] = s[t][r] * 0.125f + madd;
    }
    // ---- online softmax: rows at m = r + 8*g; n across lanes 0..15 of group
    float mnew[8], sc[8];
#pragma unroll
    for (int r = 0; r < 8; ++r) {
      float v = fmaxf(s[0][r], s[1][r]);
#pragma unroll
      for (int off = 1; off < 16; off <<= 1) v = fmaxf(v, __shfl_xor(v, off, 32));
      mnew[r] = fmaxf(mrow[r], v);
      sc[r] = __expf(mrow[r] - mnew[r]);
      mrow[r] = mnew[r];
    }
#pragma unroll
    for (int r = 0; r < 8; ++r) {
      float p0 = __expf(s[0][r] - mnew[r]);
      float p1 = __expf(s[1][r] - mnew[r]);
      s[0][r] = p0; s[1][r] = p1;
      float rs = p0 + p1;
#pragma unroll
      for (int off = 1; off < 16; off <<= 1) rs += __shfl_xor(rs, off, 32);
      lrow[r] = lrow[r] * sc[r] + rs;
    }
#pragma unroll
    for (int c = 0; c < 4; ++c)
#pragma unroll
      for (int r = 0; r < 8; ++r) acc[c][r] *= sc[r];

    // ---- P (C-layout) -> per-wave LDS [16 rows][32 keys] bf16 (intra-wave)
    u16* pw = lds_p[wave];
#pragma unroll
    for (int t = 0; t < 2; ++t)
#pragma unroll
      for (int r = 0; r < 8; ++r)
        pw[(r + 8 * g) * 32 + t * 16 + m] = f2bf(s[t][r]);
    wait_ds0();
    v16bf pa;                               // A-tile from LDS (ld = 32)
    {
      const u16* base = pw + m * 32 + g * 8;
      u16x8 lo = *(const u16x8*)base;
      u16x8 hi = *(const u16x8*)(base + 16);
      BfPack rr;
#pragma unroll
      for (int i = 0; i < 8; ++i) { rr.u[i] = lo[i]; rr.u[8 + i] = hi[i]; }
      pa = rr.b;
    }

    // ---- acc += P(16x32) @ V(32x64), B-operand from staged Vt tile
#pragma unroll
    for (int c = 0; c < 4; ++c) {
      v16bf vb = load_b32x16(vtile, c * 16, 0, 32, lane);
      acc[c] = wmma_bf16(pa, vb, acc[c]);
    }
  }

  // ---- normalize and emit to [B][S][H*DK] bf16 for the output projection
  float rinv[8];
#pragma unroll
  for (int r = 0; r < 8; ++r) rinv[r] = 1.0f / lrow[r];
#pragma unroll
  for (int c = 0; c < 4; ++c)
#pragma unroll
    for (int r = 0; r < 8; ++r) {
      int row = q0 + r + 8 * g;
      int col = h * DK + c * 16 + m;
      attn_out[((size_t)b * S_LEN + row) * DMODEL + col] = f2bf(acc[c][r] * rinv[r]);
    }
}

// ---------------- output projection ----------------
// attn:[4096 x 1024] bf16, woT:[1024 x 1024] bf16 (W_out^T) -> f32 out
__global__ __launch_bounds__(32) void out_gemm_kernel(
    const u16* __restrict__ attn, const u16* __restrict__ woT,
    const float* __restrict__ bout, float* __restrict__ out) {
  int lane = threadIdx.x;
  int ntile = blockIdx.x & (DMODEL / 128 - 1);   // 8
  int mtile = blockIdx.x >> 3;
  int m0 = mtile * 16, n0 = ntile * 128;
  v8f acc[8];
#pragma unroll
  for (int c = 0; c < 8; ++c) acc[c] = 0.0f;
  gemm_tile_16x128(attn, woT, DMODEL, m0, n0, lane, acc);

  int m = lane & 15, g = lane >> 4;
#pragma unroll
  for (int c = 0; c < 8; ++c) {
    int ncol = n0 + c * 16 + m;
    float bias = bout[ncol];
#pragma unroll
    for (int r = 0; r < 8; ++r) {
      int row = m0 + r + 8 * g;
      out[(size_t)row * DMODEL + ncol] = acc[c][r] + bias;
    }
  }
}

// ---------------- host launcher ----------------

extern "C" void kernel_launch(void* const* d_in, const int* in_sizes, int n_in,
                              void* d_out, int out_size, void* d_ws, size_t ws_size,
                              hipStream_t stream) {
  const float* x    = (const float*)d_in[0];
  const int*   mask = (const int*)d_in[1];
  const float* Wqkv = (const float*)d_in[2];
  const float* bqkv = (const float*)d_in[3];
  const float* Wout = (const float*)d_in[4];
  const float* bout = (const float*)d_in[5];
  float* out = (float*)d_out;

  const int M = BATCH * S_LEN;                 // 4096
  char* ws = (char*)d_ws;
  const size_t SZ_X   = (size_t)M * DMODEL * 2;          // x bf16
  const size_t SZ_WQ  = (size_t)3 * DMODEL * DMODEL * 2; // Wqkv^T bf16
  const size_t SZ_WO  = (size_t)DMODEL * DMODEL * 2;     // Wout^T bf16
  const size_t SZ_HD  = (size_t)BATCH * NHEADS * S_LEN * DK * 2; // q/k/vt each
  u16* xb  = (u16*)(ws);
  u16* wqT = (u16*)(ws + SZ_X);
  u16* woT = (u16*)(ws + SZ_X + SZ_WQ);
  u16* qb  = (u16*)(ws + SZ_X + SZ_WQ + SZ_WO);
  u16* kb  = (u16*)(ws + SZ_X + SZ_WQ + SZ_WO + SZ_HD);
  u16* vtb = (u16*)(ws + SZ_X + SZ_WQ + SZ_WO + 2 * SZ_HD);
  u16* ab  = (u16*)(ws + SZ_X + SZ_WQ + SZ_WO + 3 * SZ_HD);

  int n_x = M * DMODEL;
  cast_bf16_kernel<<<(n_x + 255) / 256, 256, 0, stream>>>(x, xb, n_x);
  int n_wq = DMODEL * 3 * DMODEL;
  transpose_cast_kernel<<<(n_wq + 255) / 256, 256, 0, stream>>>(Wqkv, wqT, DMODEL,
                                                                3 * DMODEL);
  int n_wo = DMODEL * DMODEL;
  transpose_cast_kernel<<<(n_wo + 255) / 256, 256, 0, stream>>>(Wout, woT, DMODEL,
                                                                DMODEL);

  qkv_gemm_kernel<<<dim3((M / 16) * (3 * DMODEL / 128)), 32, 0, stream>>>(
      xb, wqT, bqkv, qb, kb, vtb);

  attn_kernel<<<dim3(BATCH * NHEADS * (S_LEN / 64)), 128, 0, stream>>>(
      qb, kb, vtb, mask, ab);

  out_gemm_kernel<<<dim3((M / 16) * (DMODEL / 128)), 32, 0, stream>>>(
      ab, woT, bout, out);
}
